// LSTM_GAT_Model_53128745451598
// MI455X (gfx1250) — compile-verified, matrix operand order checked by reading, not run on hardware
//
#include <hip/hip_runtime.h>
#include <math.h>

// Model constants (from the reference)
#define T_SEQ 32
#define IN_F  32
#define HID   64
#define NGATE 256   // 4*H gate width

typedef __attribute__((ext_vector_type(16))) _Float16 v16h;
typedef __attribute__((ext_vector_type(8)))  _Float16 v8h;
typedef __attribute__((ext_vector_type(8)))  float    v8f;

union AFrag { v16h v; v8h h[2]; };

__device__ __forceinline__ v8f wmma16(v16h a, v16h b, v8f c) {
  // D = A(16x32 f16) * B(32x16 f16) + C(16x16 f32)
  return __builtin_amdgcn_wmma_f32_16x16x32_f16(false, a, false, b, (short)0, c,
                                                false, false);
}

// gfx1250 hardware tanh (trans op). v_nop covers the TRANS co-exec hazard
// (1 independent op needed before the result is consumed / sources reused).
__device__ __forceinline__ float htanh(float x) {
  float r;
  asm("v_tanh_f32 %0, %1\n\tv_nop" : "=v"(r) : "v"(x));
  return r;
}
__device__ __forceinline__ float hsig(float x) { return 0.5f + 0.5f * htanh(0.5f * x); }

// A-fragment from a row-major f16 tile (LDS): lane holds row (lane&15),
// lanes<16 take K {base..base+7, base+16..base+23}, lanes>=16 the mirror.
__device__ __forceinline__ v16h load_a_lds(const _Float16* base, int stride,
                                           int lane, int kbase) {
  const int r = lane & 15;
  const int s = (lane >> 4) << 3;
  AFrag u;
  u.h[0] = *(const v8h*)(base + r * stride + kbase + s);
  u.h[1] = *(const v8h*)(base + r * stride + kbase + 16 + s);
  return u.v;
}

// B-fragment pre-packed fragment-major: 32 lanes x 16 halfs contiguous.
__device__ __forceinline__ v16h load_b_lds(const _Float16* frag, int lane) {
  AFrag u;
  const v8h* p = (const v8h*)(frag + lane * 16);
  u.h[0] = p[0];
  u.h[1] = p[1];
  return u.v;
}

// Build an f16 A-fragment from 16 prefetched f32 values (x streaming path).
__device__ __forceinline__ v16h cvt16(float4 a, float4 b, float4 c, float4 d) {
  v16h r;
  r[0]  = (_Float16)a.x; r[1]  = (_Float16)a.y; r[2]  = (_Float16)a.z; r[3]  = (_Float16)a.w;
  r[4]  = (_Float16)b.x; r[5]  = (_Float16)b.y; r[6]  = (_Float16)b.z; r[7]  = (_Float16)b.w;
  r[8]  = (_Float16)c.x; r[9]  = (_Float16)c.y; r[10] = (_Float16)c.z; r[11] = (_Float16)c.w;
  r[12] = (_Float16)d.x; r[13] = (_Float16)d.y; r[14] = (_Float16)d.z; r[15] = (_Float16)d.w;
  return r;
}

// ---------------------------------------------------------------------------
// Kernel 1: pack weights into f16 WMMA B-fragments + fuse biases.
// Fragment order: W0ih[16] W0hh[32] W1ih[32] W1hh[32] G1w[32]  (512 halfs each)
// ---------------------------------------------------------------------------
#define PACK_HALFS (144 * 512)   // 73728

__global__ void prep_pack(const float* __restrict__ w0ih, const float* __restrict__ w0hh,
                          const float* __restrict__ w1ih, const float* __restrict__ w1hh,
                          const float* __restrict__ g1w,
                          const float* __restrict__ b0i, const float* __restrict__ b0h,
                          const float* __restrict__ b1i, const float* __restrict__ b1h,
                          _Float16* __restrict__ wpack, float* __restrict__ bias) {
  int p = blockIdx.x * blockDim.x + threadIdx.x;
  if (p < PACK_HALFS) {
    int frag = p >> 9, r = p & 511;
    int lane = r >> 4, i = r & 15;
    const float* W; int Kd; bool tr; int lf;
    if      (frag < 16)  { W = w0ih; Kd = 32; tr = true;  lf = frag;       }
    else if (frag < 48)  { W = w0hh; Kd = 64; tr = true;  lf = frag - 16;  }
    else if (frag < 80)  { W = w1ih; Kd = 64; tr = true;  lf = frag - 48;  }
    else if (frag < 112) { W = w1hh; Kd = 64; tr = true;  lf = frag - 80;  }
    else                 { W = g1w;  Kd = 64; tr = false; lf = frag - 112; }
    int kc = lf >> 4, nt = lf & 15;
    int n = nt * 16 + (lane & 15);
    int kk = (i & 7) + ((i >> 3) << 4) + ((lane >> 4) << 3);
    int k = kc * 32 + kk;
    float v = tr ? W[n * Kd + k] : W[k * NGATE + n];
    wpack[p] = (_Float16)v;
  } else if (p < PACK_HALFS + 512) {
    int j = p - PACK_HALFS;
    bias[j] = (j < 256) ? (b0i[j] + b0h[j]) : (b1i[j - 256] + b1h[j - 256]);
  }
}

// ---------------------------------------------------------------------------
// Kernel 2: fused 2-layer LSTM.
// 4 waves/WG, 2 node-tiles (32 nodes) per wave: every B fragment feeds 2 WMMAs
// (halves LDS B-traffic per FLOP, 8 independent accum chains per jt for ILP).
// x is streamed from global with 1-step prefetch; h tiles live in LDS.
// ---------------------------------------------------------------------------
#define SMEM_W_BYTES  114688                       // 112 frags * 1KB
#define SMEM_BIAS_OFF 114688                       // 512 f32 biases
#define SMEM_WAVE_OFF 116736
#define SMEM_WAVE_SZ  16384                        // h0 (2 tiles x 2 buf) + h1
#define SMEM_TOTAL    (SMEM_WAVE_OFF + 4 * SMEM_WAVE_SZ)   // 182272 < 320KB

__global__ __launch_bounds__(128, 1) void lstm_fused(
    const float* __restrict__ x, const _Float16* __restrict__ wpack,
    const float* __restrict__ bias, _Float16* __restrict__ hfin16, int Nn) {
  extern __shared__ __align__(16) char smem[];
  const int tid = threadIdx.x, wid = tid >> 5, lane = tid & 31;

  { // cooperative weight + bias load into LDS (shared by all 4 waves)
    const int4* g = (const int4*)wpack;
    int4* s = (int4*)smem;
    for (int i = tid; i < SMEM_W_BYTES / 16; i += blockDim.x) s[i] = g[i];
    float* sb = (float*)(smem + SMEM_BIAS_OFF);
    for (int i = tid; i < 512; i += blockDim.x) sb[i] = bias[i];
  }
  const _Float16* sW = (const _Float16*)smem;
  const float*    sB = (const float*)(smem + SMEM_BIAS_OFF);

  // per-wave LDS: h0[tile][buf][16][64], h1[tile][buf][16][64]  (f16)
  _Float16* h0 = (_Float16*)(smem + SMEM_WAVE_OFF + wid * SMEM_WAVE_SZ);
  _Float16* h1 = h0 + 4096;

  const int node_base = (blockIdx.x * 4 + wid) * 32;   // 2 tiles of 16 nodes

  for (int i = lane; i < 8192; i += 32) h0[i] = (_Float16)0.0f; // zeros h0+h1
  __syncthreads();

  float c0[2][4][8], c1[2][4][8];
#pragma unroll
  for (int tl = 0; tl < 2; ++tl)
#pragma unroll
    for (int a = 0; a < 4; ++a)
#pragma unroll
      for (int b = 0; b < 8; ++b) { c0[tl][a][b] = 0.0f; c1[tl][a][b] = 0.0f; }

  int cur0 = 0, cur1 = 0;
  const int lcol = lane & 15;
  const int mrow = (lane >> 4) << 3;

  // x prefetch registers: 16 f32 per tile per step
  float4 xr[2][4];
  auto load_x = [&](int t) {
#pragma unroll
    for (int tl = 0; tl < 2; ++tl) {
      int node = min(node_base + tl * 16 + lcol, Nn - 1);
      const float4* xp =
          (const float4*)(x + ((size_t)node * T_SEQ + t) * IN_F + ((lane >> 4) << 3));
      __builtin_prefetch(xp + 8, 0, 1);    // next timestep's row
      xr[tl][0] = xp[0];
      xr[tl][1] = xp[1];
      xr[tl][2] = xp[4];
      xr[tl][3] = xp[5];
    }
  };
  load_x(0);

#pragma unroll 1
  for (int t = 0; t < T_SEQ; ++t) {
    // ---------------- layer 0 ----------------
    v16h axf[2];
#pragma unroll
    for (int tl = 0; tl < 2; ++tl)
      axf[tl] = cvt16(xr[tl][0], xr[tl][1], xr[tl][2], xr[tl][3]);
    load_x(min(t + 1, T_SEQ - 1));         // prefetch next step (branchless)

    v16h a0[2][2];
#pragma unroll
    for (int tl = 0; tl < 2; ++tl) {
      const _Float16* hp = h0 + tl * 2048 + cur0 * 1024;
      a0[tl][0] = load_a_lds(hp, 64, lane, 0);
      a0[tl][1] = load_a_lds(hp, 64, lane, 32);
    }

#pragma unroll
    for (int jt = 0; jt < 4; ++jt) {
      v16h b[12];                          // batch all B frags for this jt
#pragma unroll
      for (int g = 0; g < 4; ++g) {
        int nt = g * 4 + jt;
        b[3 * g + 0] = load_b_lds(sW + (size_t)nt * 512, lane);
        b[3 * g + 1] = load_b_lds(sW + (size_t)(16 + nt) * 512, lane);
        b[3 * g + 2] = load_b_lds(sW + (size_t)(32 + nt) * 512, lane);
      }
      v8f acc[2][4];
#pragma unroll
      for (int g = 0; g < 4; ++g)
#pragma unroll
        for (int tl = 0; tl < 2; ++tl) {
          v8f a = {};
          a = wmma16(axf[tl],   b[3 * g + 0], a);
          a = wmma16(a0[tl][0], b[3 * g + 1], a);
          a = wmma16(a0[tl][1], b[3 * g + 2], a);
          acc[tl][g] = a;
        }
      int j = jt * 16 + lcol;
      float bI = sB[j], bF = sB[64 + j], bG = sB[128 + j], bO = sB[192 + j];
#pragma unroll
      for (int tl = 0; tl < 2; ++tl) {
        _Float16* h0n = h0 + tl * 2048 + (cur0 ^ 1) * 1024;
#pragma unroll
        for (int v = 0; v < 8; ++v) {
          float ig = hsig(acc[tl][0][v] + bI);
          float fg = hsig(acc[tl][1][v] + bF);
          float gg = htanh(acc[tl][2][v] + bG);
          float og = hsig(acc[tl][3][v] + bO);
          float c = fg * c0[tl][jt][v] + ig * gg;
          c0[tl][jt][v] = c;
          h0n[(mrow + v) * 64 + j] = (_Float16)(og * htanh(c));
        }
      }
    }
    cur0 ^= 1;
    asm volatile("s_wait_dscnt 0x0" ::: "memory");

    // ---------------- layer 1 (input = h0 just written) ----------------
    v16h in0[2], in1[2], r0[2], r1[2];
#pragma unroll
    for (int tl = 0; tl < 2; ++tl) {
      const _Float16* ip = h0 + tl * 2048 + cur0 * 1024;
      const _Float16* rp = h1 + tl * 2048 + cur1 * 1024;
      in0[tl] = load_a_lds(ip, 64, lane, 0);
      in1[tl] = load_a_lds(ip, 64, lane, 32);
      r0[tl]  = load_a_lds(rp, 64, lane, 0);
      r1[tl]  = load_a_lds(rp, 64, lane, 32);
    }

#pragma unroll
    for (int jt = 0; jt < 4; ++jt) {
      v16h b[16];
#pragma unroll
      for (int g = 0; g < 4; ++g) {
        int nt = g * 4 + jt;
        b[4 * g + 0] = load_b_lds(sW + (size_t)(48 + nt) * 512, lane);
        b[4 * g + 1] = load_b_lds(sW + (size_t)(64 + nt) * 512, lane);
        b[4 * g + 2] = load_b_lds(sW + (size_t)(80 + nt) * 512, lane);
        b[4 * g + 3] = load_b_lds(sW + (size_t)(96 + nt) * 512, lane);
      }
      v8f acc[2][4];
#pragma unroll
      for (int g = 0; g < 4; ++g)
#pragma unroll
        for (int tl = 0; tl < 2; ++tl) {
          v8f a = {};
          a = wmma16(in0[tl], b[4 * g + 0], a);
          a = wmma16(in1[tl], b[4 * g + 1], a);
          a = wmma16(r0[tl],  b[4 * g + 2], a);
          a = wmma16(r1[tl],  b[4 * g + 3], a);
          acc[tl][g] = a;
        }
      int j = jt * 16 + lcol;
      float bI = sB[256 + j], bF = sB[320 + j], bG = sB[384 + j], bO = sB[448 + j];
#pragma unroll
      for (int tl = 0; tl < 2; ++tl) {
        _Float16* h1n = h1 + tl * 2048 + (cur1 ^ 1) * 1024;
#pragma unroll
        for (int v = 0; v < 8; ++v) {
          float ig = hsig(acc[tl][0][v] + bI);
          float fg = hsig(acc[tl][1][v] + bF);
          float gg = htanh(acc[tl][2][v] + bG);
          float og = hsig(acc[tl][3][v] + bO);
          float c = fg * c1[tl][jt][v] + ig * gg;
          c1[tl][jt][v] = c;
          h1n[(mrow + v) * 64 + j] = (_Float16)(og * htanh(c));
        }
      }
    }
    cur1 ^= 1;
    asm volatile("s_wait_dscnt 0x0" ::: "memory");
  }

  // final hidden state (f16) feeds the GAT1 WMMA GEMM
#pragma unroll
  for (int tl = 0; tl < 2; ++tl) {
    const _Float16* hf = h1 + tl * 2048 + cur1 * 1024;
    for (int i = lane; i < 1024; i += 32) {
      int r = i >> 6, cc = i & 63;
      int node = node_base + tl * 16 + r;
      if (node < Nn) hfin16[(size_t)node * 64 + cc] = hf[i];
    }
  }
}

// ---------------------------------------------------------------------------
// Kernel 3: GAT1 feature GEMM  xh = h(Nx64) @ g1_w(64x256), WMMA.
// ---------------------------------------------------------------------------
__global__ __launch_bounds__(128) void gat1_gemm(
    const _Float16* __restrict__ hfin16, const _Float16* __restrict__ g1pack,
    float* __restrict__ xh1, int Nn) {
  __shared__ _Float16 sW[32 * 512];   // 32 KB of packed B fragments
  const int tid = threadIdx.x, wid = tid >> 5, lane = tid & 31;
  {
    const int4* g = (const int4*)g1pack;
    int4* s = (int4*)sW;
    for (int i = tid; i < 2048; i += blockDim.x) s[i] = g[i];
  }
  __syncthreads();

  const int tile = blockIdx.x * 4 + wid;
  const int node_base = tile * 16;
  const int node = min(node_base + (lane & 15), Nn - 1);
  const _Float16* hp = hfin16 + (size_t)node * 64 + ((lane >> 4) << 3);
  AFrag a0, a1;
  a0.h[0] = *(const v8h*)(hp);
  a0.h[1] = *(const v8h*)(hp + 16);
  a1.h[0] = *(const v8h*)(hp + 32);
  a1.h[1] = *(const v8h*)(hp + 48);
  const int lcol = lane & 15, mrow = (lane >> 4) << 3;
#pragma unroll
  for (int nt = 0; nt < 16; ++nt) {
    v8f acc = {};
    acc = wmma16(a0.v, load_b_lds(sW + nt * 512, lane), acc);
    acc = wmma16(a1.v, load_b_lds(sW + (16 + nt) * 512, lane), acc);
    int col = nt * 16 + lcol;
#pragma unroll
    for (int v = 0; v < 8; ++v) {
      int n2 = node_base + mrow + v;
      if (n2 < Nn) xh1[(size_t)n2 * NGATE + col] = acc[v];
    }
  }
}

// ---------------------------------------------------------------------------
// GAT helper kernels
// ---------------------------------------------------------------------------
__global__ void init_buf(float* __restrict__ out1, float* __restrict__ m1,
                         float* __restrict__ den1, float* __restrict__ m2,
                         float* __restrict__ den2, float* __restrict__ dout, int Nn) {
  int i = blockIdx.x * blockDim.x + threadIdx.x;
  if (i < Nn * NGATE) out1[i] = 0.0f;
  if (i < Nn * 4) { m1[i] = -3.402823466e38f; den1[i] = 0.0f; }
  if (i < Nn)     { m2[i] = -3.402823466e38f; den2[i] = 0.0f; dout[i] = 0.0f; }
}

// wave-per-(node,head) attention logits: al = sum_j xh[n,h,j]*a[h,j]
__global__ void gat1_al(const float* __restrict__ xh1, const float* __restrict__ as,
                        const float* __restrict__ ad, float* __restrict__ als,
                        float* __restrict__ ald, int Nn) {
  int gw = (blockIdx.x * blockDim.x + threadIdx.x) >> 5;
  int lane = threadIdx.x & 31;
  if (gw >= Nn * 4) return;
  int n = gw >> 2, h = gw & 3;
  const float* xr = xh1 + (size_t)n * NGATE + h * 64;
  float ps = xr[lane] * as[h * 64 + lane] + xr[lane + 32] * as[h * 64 + lane + 32];
  float pd = xr[lane] * ad[h * 64 + lane] + xr[lane + 32] * ad[h * 64 + lane + 32];
#pragma unroll
  for (int off = 16; off > 0; off >>= 1) {
    ps += __shfl_xor(ps, off, 32);
    pd += __shfl_xor(pd, off, 32);
  }
  if (lane == 0) { als[gw] = ps; ald[gw] = pd; }
}

__device__ __forceinline__ void atomicMaxF(float* addr, float v) {
  if (v >= 0.0f) atomicMax((int*)addr, __float_as_int(v));
  else           atomicMin((unsigned int*)addr, __float_as_uint(v));
}

template<int HH>
__global__ void edge_max(const int* __restrict__ src, const int* __restrict__ dst,
                         int E, int Nn, const float* __restrict__ als,
                         const float* __restrict__ ald, float* __restrict__ m) {
  int tid = blockIdx.x * blockDim.x + threadIdx.x;
  if (tid >= (E + Nn) * HH) return;
  int slot = tid / HH, h = tid - slot * HH;
  int s = slot < E ? src[slot] : slot - E;
  int d = slot < E ? dst[slot] : slot - E;
  float e = als[s * HH + h] + ald[d * HH + h];
  e = e > 0.0f ? e : 0.2f * e;
  atomicMaxF(&m[d * HH + h], e);
}

template<int HH>
__global__ void edge_den(const int* __restrict__ src, const int* __restrict__ dst,
                         int E, int Nn, const float* __restrict__ als,
                         const float* __restrict__ ald, const float* __restrict__ m,
                         float* __restrict__ den) {
  int tid = blockIdx.x * blockDim.x + threadIdx.x;
  if (tid >= (E + Nn) * HH) return;
  int slot = tid / HH, h = tid - slot * HH;
  int s = slot < E ? src[slot] : slot - E;
  int d = slot < E ? dst[slot] : slot - E;
  float e = als[s * HH + h] + ald[d * HH + h];
  e = e > 0.0f ? e : 0.2f * e;
  float mm = m[d * HH + h]; if (mm < -1e37f) mm = 0.0f;
  atomicAdd(&den[d * HH + h], __expf(e - mm));
}

template<int HH, int F>
__global__ void edge_agg(const int* __restrict__ src, const int* __restrict__ dst,
                         int E, int Nn, const float* __restrict__ als,
                         const float* __restrict__ ald, const float* __restrict__ m,
                         const float* __restrict__ den, const float* __restrict__ xh,
                         float* __restrict__ out) {
  int tid = blockIdx.x * blockDim.x + threadIdx.x;
  if (tid >= (E + Nn) * HH) return;
  int slot = tid / HH, h = tid - slot * HH;
  int s = slot < E ? src[slot] : slot - E;
  int d = slot < E ? dst[slot] : slot - E;
  float e = als[s * HH + h] + ald[d * HH + h];
  e = e > 0.0f ? e : 0.2f * e;
  float mm = m[d * HH + h]; if (mm < -1e37f) mm = 0.0f;
  float w = __expf(e - mm) / (den[d * HH + h] + 1e-16f);
  const float* xr = xh + (size_t)s * (HH * F) + h * F;
  float* orow     = out + (size_t)d * (HH * F) + h * F;
#pragma unroll 4
  for (int c = 0; c < F; ++c) atomicAdd(&orow[c], xr[c] * w);
}

__global__ void relu_bias(float* __restrict__ out1, const float* __restrict__ b, int Nn) {
  int i = blockIdx.x * blockDim.x + threadIdx.x;
  if (i < Nn * NGATE) {
    float v = out1[i] + b[i & 255];
    out1[i] = v > 0.0f ? v : 0.0f;
  }
}

// wave-per-node 256-wide dot:  xh2 = h1out @ g2_w, plus attention logits
__global__ void gat2_dot(const float* __restrict__ h1o, const float* __restrict__ w,
                         const float* __restrict__ as, const float* __restrict__ ad,
                         float* __restrict__ xh2, float* __restrict__ als2,
                         float* __restrict__ ald2, int Nn) {
  int gw = (blockIdx.x * blockDim.x + threadIdx.x) >> 5;
  int lane = threadIdx.x & 31;
  if (gw >= Nn) return;
  const float* r = h1o + (size_t)gw * NGATE;
  float p = 0.0f;
#pragma unroll
  for (int ii = 0; ii < 8; ++ii) p += r[lane + 32 * ii] * w[lane + 32 * ii];
#pragma unroll
  for (int off = 16; off > 0; off >>= 1) p += __shfl_xor(p, off, 32);
  if (lane == 0) { xh2[gw] = p; als2[gw] = p * as[0]; ald2[gw] = p * ad[0]; }
}

__global__ void add_bias(float* __restrict__ dout, const float* __restrict__ b, int Nn) {
  int i = blockIdx.x * blockDim.x + threadIdx.x;
  if (i < Nn) dout[i] += b[0];
}

// ---------------------------------------------------------------------------
extern "C" void kernel_launch(void* const* d_in, const int* in_sizes, int n_in,
                              void* d_out, int out_size, void* d_ws, size_t ws_size,
                              hipStream_t stream) {
  const float* x    = (const float*)d_in[0];
  const int*   ei   = (const int*)  d_in[1];
  const float* w0ih = (const float*)d_in[2];
  const float* w0hh = (const float*)d_in[3];
  const float* b0i  = (const float*)d_in[4];
  const float* b0h  = (const float*)d_in[5];
  const float* w1ih = (const float*)d_in[6];
  const float* w1hh = (const float*)d_in[7];
  const float* b1i  = (const float*)d_in[8];
  const float* b1h  = (const float*)d_in[9];
  const float* g1w  = (const float*)d_in[10];
  const float* g1as = (const float*)d_in[11];
  const float* g1ad = (const float*)d_in[12];
  const float* g1b  = (const float*)d_in[13];
  const float* g2w  = (const float*)d_in[14];
  const float* g2as = (const float*)d_in[15];
  const float* g2ad = (const float*)d_in[16];
  const float* g2b  = (const float*)d_in[17];
  float* dout = (float*)d_out;

  const int Nn = in_sizes[0] / (T_SEQ * IN_F);
  const int E  = in_sizes[1] / 2;
  const int* src = ei;
  const int* dst = ei + E;

  char* ws = (char*)d_ws;
  size_t off = 0;
  auto take = [&](size_t bytes) {
    char* p = ws + off;
    off = (off + bytes + 255) & ~(size_t)255;
    return p;
  };
  _Float16* wpack  = (_Float16*)take((size_t)PACK_HALFS * 2);
  float*    bias   = (float*)   take(512 * 4);
  _Float16* hfin16 = (_Float16*)take((size_t)Nn * 64 * 2);
  float*    xh1    = (float*)   take((size_t)Nn * NGATE * 4);
  float*    out1   = (float*)   take((size_t)Nn * NGATE * 4);
  float*    als1   = (float*)   take((size_t)Nn * 4 * 4);
  float*    ald1   = (float*)   take((size_t)Nn * 4 * 4);
  float*    m1     = (float*)   take((size_t)Nn * 4 * 4);
  float*    den1   = (float*)   take((size_t)Nn * 4 * 4);
  float*    xh2    = (float*)   take((size_t)Nn * 4);
  float*    als2   = (float*)   take((size_t)Nn * 4);
  float*    ald2   = (float*)   take((size_t)Nn * 4);
  float*    m2     = (float*)   take((size_t)Nn * 4);
  float*    den2   = (float*)   take((size_t)Nn * 4);
  (void)ws_size; (void)n_in; (void)out_size;

  // 1. pack weights (f16 WMMA fragments) + fused biases
  prep_pack<<<(PACK_HALFS + 512 + 255) / 256, 256, 0, stream>>>(
      w0ih, w0hh, w1ih, w1hh, g1w, b0i, b0h, b1i, b1h, wpack, bias);

  // 2. zero/neg-init all accumulators (d_out included) every call
  init_buf<<<((size_t)Nn * NGATE + 255) / 256, 256, 0, stream>>>(
      out1, m1, den1, m2, den2, dout, Nn);

  // 3. fused 2-layer LSTM (WMMA, LDS-resident recurrence, dual tile/wave)
  hipFuncSetAttribute(reinterpret_cast<const void*>(&lstm_fused),
                      hipFuncAttributeMaxDynamicSharedMemorySize, SMEM_TOTAL);
  const int tiles = (Nn + 15) / 16;                 // 16-node tiles
  const int lstm_wgs = (tiles + 7) / 8;             // 8 tiles per WG
  lstm_fused<<<lstm_wgs, 128, SMEM_TOTAL, stream>>>(x, wpack, bias, hfin16, Nn);

  // 4. GAT1 feature GEMM (WMMA)
  gat1_gemm<<<(tiles + 3) / 4, 128, 0, stream>>>(hfin16, wpack + 112 * 512, xh1, Nn);

  // 5. GAT1 attention + softmax-scatter (implicit self loops)
  gat1_al<<<((Nn * 4) + 7) / 8, 256, 0, stream>>>(xh1, g1as, g1ad, als1, ald1, Nn);
  const int ET = E + Nn;
  edge_max<4><<<((size_t)ET * 4 + 255) / 256, 256, 0, stream>>>(src, dst, E, Nn, als1, ald1, m1);
  edge_den<4><<<((size_t)ET * 4 + 255) / 256, 256, 0, stream>>>(src, dst, E, Nn, als1, ald1, m1, den1);
  edge_agg<4, 64><<<((size_t)ET * 4 + 255) / 256, 256, 0, stream>>>(
      src, dst, E, Nn, als1, ald1, m1, den1, xh1, out1);
  relu_bias<<<((size_t)Nn * NGATE + 255) / 256, 256, 0, stream>>>(out1, g1b, Nn);

  // 6. GAT2 (256 -> 1) + softmax-scatter into d_out
  gat2_dot<<<(Nn + 7) / 8, 256, 0, stream>>>(out1, g2w, g2as, g2ad, xh2, als2, ald2, Nn);
  edge_max<1><<<(ET + 255) / 256, 256, 0, stream>>>(src, dst, E, Nn, als2, ald2, m2);
  edge_den<1><<<(ET + 255) / 256, 256, 0, stream>>>(src, dst, E, Nn, als2, ald2, m2, den2);
  edge_agg<1, 1><<<(ET + 255) / 256, 256, 0, stream>>>(
      src, dst, E, Nn, als2, ald2, m2, den2, xh2, dout);
  add_bias<<<(Nn + 255) / 256, 256, 0, stream>>>(dout, g2b, Nn);
}
